// MSDeformAttn_80144089743527
// MI455X (gfx1250) — matrix operand order repaired and checked by reading, not compiled, not run on hardware
//
#include <hip/hip_runtime.h>
#include <hip/hip_bf16.h>

typedef __bf16 bf16;
typedef __attribute__((ext_vector_type(16))) __bf16 v16bf;
typedef __attribute__((ext_vector_type(8)))  __bf16 v8bf;
typedef __attribute__((ext_vector_type(8)))  float  v8f;

#define AST 36   // A LDS row stride in floats (32 K + 4 pad) = 144 B, keeps 16B alignment
#define BST 40   // B LDS row stride in bf16  (32 K + 8 pad) =  80 B, keeps 16B alignment

__device__ __forceinline__ v16bf cat8(v8bf lo, v8bf hi) {
  return __builtin_shufflevector(lo, hi, 0,1,2,3,4,5,6,7,8,9,10,11,12,13,14,15);
}

// 16-byte async copy global -> LDS (gfx1250, tracked by ASYNCcnt, no VGPR data).
__device__ __forceinline__ void async_cp16(unsigned lds_off, const void* gaddr) {
  asm volatile("global_load_async_to_lds_b128 %0, %1, off"
               :: "v"(lds_off), "v"(gaddr) : "memory");
}
__device__ __forceinline__ void wait_async0() {
  asm volatile("s_wait_asynccnt 0x0" ::: "memory");
}
__device__ __forceinline__ unsigned lds_addr(const void* p) {
  // LDS aperture: low 32 bits of a generic LDS pointer are the LDS byte address.
  return (unsigned)(uintptr_t)p;
}

// A fragment: 16x32 bf16 per ISA 7.12.2. Source tile is f32 in LDS; convert on the fly
// (v_cvt_pk_bf16_f32 co-executes with XDL WMMA).
// lanes 0-15 (hf=0): row r, K{0..7, 16..23}; lanes 16-31 (hf=1): K{8..15, 24..31}.
__device__ __forceinline__ v16bf frag_a_f32(const float* __restrict__ As, int r, int hf) {
  const float* p = As + r * AST;
  float4 q0 = *(const float4*)(p + hf * 8);
  float4 q1 = *(const float4*)(p + hf * 8 + 4);
  float4 q2 = *(const float4*)(p + 16 + hf * 8);
  float4 q3 = *(const float4*)(p + 20 + hf * 8);
  v16bf a;
  a[0]  = (bf16)q0.x; a[1]  = (bf16)q0.y; a[2]  = (bf16)q0.z; a[3]  = (bf16)q0.w;
  a[4]  = (bf16)q1.x; a[5]  = (bf16)q1.y; a[6]  = (bf16)q1.z; a[7]  = (bf16)q1.w;
  a[8]  = (bf16)q2.x; a[9]  = (bf16)q2.y; a[10] = (bf16)q2.z; a[11] = (bf16)q2.w;
  a[12] = (bf16)q3.x; a[13] = (bf16)q3.y; a[14] = (bf16)q3.z; a[15] = (bf16)q3.w;
  return a;
}

// B fragment: 32x16 bf16, LDS layout Bs[col][k] (pre-transposed bf16 weights).
__device__ __forceinline__ v16bf frag_b_bf16(const bf16* __restrict__ Bs, int c, int hf) {
  v8bf lo = *(const v8bf*)(&Bs[c * BST + hf * 8]);
  v8bf hi = *(const v8bf*)(&Bs[c * BST + 16 + hf * 8]);
  return cat8(lo, hi);
}

// Issue async fills for one 128x32 A tile (f32) and one 128x32 Bt tile (bf16).
// OOB rows/cols are clamped: they only feed C rows/cols that are never stored.
__device__ __forceinline__ void load_tiles_async(
    const float* __restrict__ A, const bf16* __restrict__ Bt,
    int M, int N, int K, int mBase, int nBase, int k0,
    float* __restrict__ As, bf16* __restrict__ Bs, int tid)
{
  // A: 128 rows x 32 f32 = 1024 x 16B chunks (8 per row)
  for (int c = tid; c < 1024; c += 256) {
    int row = c >> 3, ch = c & 7;
    int gr = mBase + row; if (gr >= M) gr = M - 1;
    async_cp16(lds_addr(As + row * AST + ch * 4),
               A + (size_t)gr * K + k0 + ch * 4);
  }
  // Bt: 128 cols x 32 bf16 = 512 x 16B chunks (4 per col)
  for (int c = tid; c < 512; c += 256) {
    int col = c >> 2, ch = c & 3;
    int gn = nBase + col; if (gn >= N) gn = N - 1;
    async_cp16(lds_addr(Bs + col * BST + ch * 8),
               Bt + (size_t)gn * K + k0 + ch * 8);
  }
}

// C[M,N] = A[M,K] @ Bt[N,K]^T + bias.  A f32, Bt bf16 (pre-transposed), C f32.
// K multiple of 32. 256 threads = 8 waves (4M x 2N), each wave a 32x64 C region.
// Double-buffered LDS filled by GLOBAL_LOAD_ASYNC_TO_LDS_B128.
// Grid: x = N tiles (fast -> A-tile sharers adjacent, A read ~once from HBM), y = M tiles.
__global__ __launch_bounds__(256)
void gemm_wmma_async(const float* __restrict__ A, const bf16* __restrict__ Bt,
                     const float* __restrict__ bias, float* __restrict__ C,
                     int M, int N, int K)
{
  __shared__ __align__(16) float As[2][128 * AST];
  __shared__ __align__(16) bf16  Bs[2][128 * BST];

  const int tid  = threadIdx.x;
  const int lane = tid & 31;
  const int wave = tid >> 5;
  const int wr   = wave >> 1;   // 0..3 -> M offset wr*32
  const int wc   = wave & 1;    // 0..1 -> N offset wc*64
  const int hf   = lane >> 4;
  const int l16  = lane & 15;

  const int nBase = blockIdx.x * 128;
  const int mBase = blockIdx.y * 128;

  v8f acc[2][4];
  #pragma unroll
  for (int i = 0; i < 2; ++i)
    #pragma unroll
    for (int j = 0; j < 4; ++j)
      acc[i][j] = {};

  const int steps = K / 32;
  load_tiles_async(A, Bt, M, N, K, mBase, nBase, 0, As[0], Bs[0], tid);

  for (int i = 0; i < steps; ++i) {
    wait_async0();        // my wave's fills done (each wave waits before signaling)
    __syncthreads();      // all waves' fills visible; prev-iter readers finished
    if (i + 1 < steps)
      load_tiles_async(A, Bt, M, N, K, mBase, nBase, (i + 1) * 32,
                       As[(i + 1) & 1], Bs[(i + 1) & 1], tid);

    const float* Ac = As[i & 1];
    const bf16*  Bc = Bs[i & 1];

    v16bf afrag[2];
    #pragma unroll
    for (int mt = 0; mt < 2; ++mt)
      afrag[mt] = frag_a_f32(Ac, wr * 32 + mt * 16 + l16, hf);
    v16bf bfrag[4];
    #pragma unroll
    for (int nt = 0; nt < 4; ++nt)
      bfrag[nt] = frag_b_bf16(Bc, wc * 64 + nt * 16 + l16, hf);

    #pragma unroll
    for (int mt = 0; mt < 2; ++mt)
      #pragma unroll
      for (int nt = 0; nt < 4; ++nt)
        acc[mt][nt] = __builtin_amdgcn_wmma_f32_16x16x32_bf16(
            false, afrag[mt], false, bfrag[nt], (short)0, acc[mt][nt], false, false);
  }

  // C/D layout: VGPR j holds (M = j + 8*hf, N = l16)
  #pragma unroll
  for (int mt = 0; mt < 2; ++mt) {
    #pragma unroll
    for (int nt = 0; nt < 4; ++nt) {
      int colg = nBase + wc * 64 + nt * 16 + l16;
      if (colg >= N) continue;
      float bb = bias[colg];
      #pragma unroll
      for (int j = 0; j < 8; ++j) {
        int rowg = mBase + wr * 32 + mt * 16 + j + hf * 8;
        if (rowg < M) C[(size_t)rowg * N + colg] = acc[mt][nt][j] + bb;
      }
    }
  }
}

// Wt[n][k] = (bf16) W[k][n]  — weights are tiny; one-shot pre-pass.
__global__ __launch_bounds__(256)
void transpose_to_bf16(const float* __restrict__ W, bf16* __restrict__ Wt, int K, int N)
{
  int i = blockIdx.x * blockDim.x + threadIdx.x;
  if (i >= N * K) return;
  int n = i / K, k = i - n * K;
  Wt[i] = (bf16)W[(size_t)k * N + n];
}

// In-place softmax over 24 contiguous values per (q,m) row.
__global__ __launch_bounds__(256)
void softmax24_kernel(float* __restrict__ buf, int nRows)
{
  int i = blockIdx.x * blockDim.x + threadIdx.x;
  if (i >= nRows) return;
  float* p = buf + (size_t)i * 24;
  float mx = p[0];
  #pragma unroll
  for (int j = 1; j < 24; ++j) mx = fmaxf(mx, p[j]);
  float e[24];
  float s = 0.f;
  #pragma unroll
  for (int j = 0; j < 24; ++j) { e[j] = __expf(p[j] - mx); s += e[j]; }
  float inv = 1.0f / s;
  #pragma unroll
  for (int j = 0; j < 24; ++j) p[j] = e[j] * inv;
}

// One block per (q, m); 128 threads = Dh channels. 24 trilinear samples,
// corner gathers coalesced across channels (512B lines; v is L2-resident at 154 MB).
__global__ __launch_bounds__(128)
void sample_kernel(const float* __restrict__ v, const float* __restrict__ ref,
                   const float* __restrict__ off, const float* __restrict__ aw,
                   float* __restrict__ out_pre)
{
  const int S = 25039;
  const int b = blockIdx.x;
  const int q = b / 6;
  const int m = b % 6;
  const int d = threadIdx.x;

  const float rx = ref[q * 3 + 0];
  const float ry = ref[q * 3 + 1];
  const float rz = ref[q * 3 + 2];

  float acc = 0.0f;
  #pragma unroll
  for (int t = 0; t < 2; ++t) {
    #pragma unroll
    for (int l = 0; l < 3; ++l) {
      const int   DIM   = (l == 0) ? 28 : (l == 1) ? 14 : 7;
      const int   START = (l == 0) ? 0  : (l == 1) ? 21952 : 24696;
      const float fd    = (float)DIM;
      const float invn  = 1.0f / fd;
      const float* vt = v + ((size_t)t * S + START) * 768 + m * 128 + d;
      #pragma unroll
      for (int p = 0; p < 4; ++p) {
        int samp = ((m * 2 + t) * 3 + l) * 4 + p;          // index in 144 (m-major)
        float w_a = aw[(size_t)q * 144 + samp];
        const float* o = off + (size_t)q * 432 + samp * 3;
        float x = (rx + o[0] * invn) * fd - 0.5f;          // loc*DIM - 0.5
        float y = (ry + o[1] * invn) * fd - 0.5f;
        float z = (rz + o[2] * invn) * fd - 0.5f;
        float xf = floorf(x), yf = floorf(y), zf = floorf(z);
        float fx = x - xf, fy = y - yf, fz = z - zf;
        int x0 = (int)xf, y0 = (int)yf, z0 = (int)zf;
        #pragma unroll
        for (int c = 0; c < 8; ++c) {
          int dx = c & 1, dy = (c >> 1) & 1, dz = c >> 2;
          int xi = x0 + dx, yi = y0 + dy, zi = z0 + dz;
          bool ok = (xi >= 0) & (xi < DIM) & (yi >= 0) & (yi < DIM) &
                    (zi >= 0) & (zi < DIM);
          if (ok) {
            float wgt = (dx ? fx : 1.f - fx) * (dy ? fy : 1.f - fy)
                      * (dz ? fz : 1.f - fz);
            size_t idx = ((size_t)zi * DIM + yi) * DIM + xi;
            acc += w_a * wgt * vt[idx * 768];
          }
        }
      }
    }
  }
  out_pre[(size_t)q * 768 + m * 128 + d] = acc;
}

extern "C" void kernel_launch(void* const* d_in, const int* in_sizes, int n_in,
                              void* d_out, int out_size, void* d_ws, size_t ws_size,
                              hipStream_t stream)
{
  (void)in_sizes; (void)n_in; (void)out_size; (void)ws_size;

  const float* query = (const float*)d_in[0];
  const float* refp  = (const float*)d_in[1];
  const float* value = (const float*)d_in[2];
  const float* Wv    = (const float*)d_in[3];
  const float* bv    = (const float*)d_in[4];
  const float* Woff  = (const float*)d_in[5];
  const float* boff  = (const float*)d_in[6];
  const float* Wa    = (const float*)d_in[7];
  const float* ba    = (const float*)d_in[8];
  const float* Wo    = (const float*)d_in[9];
  const float* bo    = (const float*)d_in[10];
  float* out = (float*)d_out;

  const int Lq = 4096, Dm = 768, Cin = 1408;
  const int S = 25039, T = 2;
  const int Mv = T * S;                              // 50078 value rows

  float* v_buf   = (float*)d_ws;                     // 50078*768 f32
  float* off_buf = v_buf   + (size_t)Mv * Dm;        // 4096*432 f32
  float* a_buf   = off_buf + (size_t)Lq * 432;       // 4096*144 f32 (softmax in place)
  float* pre_buf = a_buf   + (size_t)Lq * 144;       // 4096*768 f32
  bf16* WvT   = (bf16*)(pre_buf + (size_t)Lq * Dm);  // [768][1408]
  bf16* WoffT = WvT   + (size_t)Dm * Cin;            // [432][768]
  bf16* WaT   = WoffT + (size_t)432 * Dm;            // [144][768]
  bf16* WoT   = WaT   + (size_t)144 * Dm;            // [768][768]

  dim3 blk(256);

  // 0) weight pre-pass: transpose + f32->bf16 (tiny)
  { int n = Dm * Cin;  transpose_to_bf16<<<dim3((n+255)/256), blk, 0, stream>>>(Wv,   WvT,   Cin, Dm); }
  { int n = 432 * Dm;  transpose_to_bf16<<<dim3((n+255)/256), blk, 0, stream>>>(Woff, WoffT, Dm, 432); }
  { int n = 144 * Dm;  transpose_to_bf16<<<dim3((n+255)/256), blk, 0, stream>>>(Wa,   WaT,   Dm, 144); }
  { int n = Dm * Dm;   transpose_to_bf16<<<dim3((n+255)/256), blk, 0, stream>>>(Wo,   WoT,   Dm, Dm); }

  // 1) v = value @ Wv + bv   (50078 x 1408 x 768) — HBM-bound; N-tiles fastest for A reuse
  { dim3 g((Dm + 127) / 128, (Mv + 127) / 128);
    gemm_wmma_async<<<g, blk, 0, stream>>>(value, WvT, bv, v_buf, Mv, Dm, Cin); }

  // 2) off = query @ Woff + boff   (4096 x 768 x 432)
  { dim3 g((432 + 127) / 128, (Lq + 127) / 128);
    gemm_wmma_async<<<g, blk, 0, stream>>>(query, WoffT, boff, off_buf, Lq, 432, Dm); }

  // 3) a = query @ Wa + ba         (4096 x 768 x 144)
  { dim3 g((144 + 127) / 128, (Lq + 127) / 128);
    gemm_wmma_async<<<g, blk, 0, stream>>>(query, WaT, ba, a_buf, Lq, 144, Dm); }

  // 4) softmax over 24 per (q,m), in place
  { int rows = Lq * 6;
    softmax24_kernel<<<dim3((rows + 255) / 256), blk, 0, stream>>>(a_buf, rows); }

  // 5) trilinear sampling + attention aggregation -> pre (4096 x 768)
  sample_kernel<<<dim3(Lq * 6), dim3(128), 0, stream>>>(v_buf, refp, off_buf, a_buf, pre_buf);

  // 6) out = pre @ Wo + bo         (4096 x 768 x 768)
  { dim3 g((Dm + 127) / 128, (Lq + 127) / 128);
    gemm_wmma_async<<<g, blk, 0, stream>>>(pre_buf, WoT, bo, out, Lq, Dm, Dm); }
}
